// PointFormerSeg_79267916415204
// MI455X (gfx1250) — compile-verified, matrix operand order checked by reading, not run on hardware
//
#include <hip/hip_runtime.h>
#include <hip/hip_bf16.h>
#include <math.h>

// ---------------------------------------------------------------------------
// Problem sizes
// ---------------------------------------------------------------------------
#define BATCH    4
#define NPTS     16384            // N
#define SPTS     1024             // S
#define M_TOT    (BATCH * NPTS)   // 65536 flattened points
#define C_IN     771
#define K1_PAD   800              // 25 * 32
#define HID      3072             // K2 = 96 * 32
#define C_OUT    768

typedef __attribute__((ext_vector_type(16))) __bf16 v16bf;
typedef __attribute__((ext_vector_type(8)))  __bf16 v8bf;
typedef __attribute__((ext_vector_type(8)))  float  v8f;

// ---------------------------------------------------------------------------
// WMMA 16-bit operand loader (ISA 7.12.2):
//   lane 0-15 : row = lane,    K = {k..k+7, k+16..k+23}
//   lane16-31 : row = lane-16, K = {k+8..k+15, k+24..k+31}
// rowbase must already include row*ld + k + ((lane>>4)*8); 16B aligned.
// ---------------------------------------------------------------------------
static __device__ __forceinline__ v16bf load_frag(const __bf16* rowbase) {
    const v8bf* p = (const v8bf*)__builtin_assume_aligned(rowbase, 16);
    v8bf lo = p[0];
    v8bf hi = p[2];       // +16 halfs
    v16bf r;
#pragma unroll
    for (int i = 0; i < 8; ++i) { r[i] = lo[i]; r[i + 8] = hi[i]; }
    return r;
}

// ---------------------------------------------------------------------------
// Kernel 0: fold BN(+conv bias) into alpha/beta, convert weights to bf16.
//   alpha = g * rsqrt(v+eps) ; beta = alpha*(bias - mean) + be
// ---------------------------------------------------------------------------
__global__ __launch_bounds__(256)
void prep_kernel(const float* __restrict__ W1, const float* __restrict__ W2,
                 const float* __restrict__ b1, const float* __restrict__ g1,
                 const float* __restrict__ be1, const float* __restrict__ m1,
                 const float* __restrict__ v1,
                 const float* __restrict__ b2, const float* __restrict__ g2,
                 const float* __restrict__ be2, const float* __restrict__ m2,
                 const float* __restrict__ v2,
                 __bf16* __restrict__ W1b, __bf16* __restrict__ W2b,
                 float* __restrict__ alpha1, float* __restrict__ beta1,
                 float* __restrict__ alpha2, float* __restrict__ beta2) {
    const size_t nW1 = (size_t)HID * K1_PAD;       // 2,457,600
    const size_t nW2 = (size_t)C_OUT * HID;        // 2,359,296
    size_t i = (size_t)blockIdx.x * 256 + threadIdx.x;
    if (i < nW1) {
        int j = (int)(i / K1_PAD), k = (int)(i % K1_PAD);
        float v = (k < C_IN) ? W1[(size_t)j * C_IN + k] : 0.0f;
        W1b[i] = (__bf16)v;
    } else if (i < nW1 + nW2) {
        size_t t = i - nW1;
        W2b[t] = (__bf16)W2[t];
    } else if (i < nW1 + nW2 + HID) {
        int j = (int)(i - nW1 - nW2);
        float s = g1[j] * (1.0f / sqrtf(v1[j] + 1e-5f));
        alpha1[j] = s;
        beta1[j]  = s * (b1[j] - m1[j]) + be1[j];
    } else if (i < nW1 + nW2 + HID + C_OUT) {
        int j = (int)(i - nW1 - nW2 - HID);
        float s = g2[j] * (1.0f / sqrtf(v2[j] + 1e-5f));
        alpha2[j] = s;
        beta2[j]  = s * (b2[j] - m2[j]) + be2[j];
    }
}

// ---------------------------------------------------------------------------
// Kernel 1: 3-NN + inverse-distance weights. One thread per query point,
// xyz2 for the batch staged in LDS (12 KB).
// ---------------------------------------------------------------------------
__global__ __launch_bounds__(256)
void knn_kernel(const float* __restrict__ xyz1, const float* __restrict__ xyz2,
                int* __restrict__ idx3, float* __restrict__ w3) {
    __shared__ float sx[SPTS], sy[SPTS], sz[SPTS];
    const int b = blockIdx.y;
    const int tid = threadIdx.x;
    const float* x2 = xyz2 + (size_t)b * 3 * SPTS;
    for (int i = tid; i < SPTS; i += 256) {
        sx[i] = x2[i];
        sy[i] = x2[SPTS + i];
        sz[i] = x2[2 * SPTS + i];
    }
    __syncthreads();

    const int n = blockIdx.x * 256 + tid;
    const float* x1 = xyz1 + (size_t)b * 3 * NPTS;
    const float px = x1[n], py = x1[NPTS + n], pz = x1[2 * NPTS + n];

    float d0 = 3.4e38f, d1 = 3.4e38f, d2 = 3.4e38f;
    int   i0 = 0, i1 = 0, i2 = 0;
    for (int s = 0; s < SPTS; ++s) {
        float dx = px - sx[s], dy = py - sy[s], dz = pz - sz[s];
        float d = dx * dx + dy * dy + dz * dz;
        if (d < d2) {
            if (d < d1) {
                if (d < d0) { d2 = d1; i2 = i1; d1 = d0; i1 = i0; d0 = d; i0 = s; }
                else        { d2 = d1; i2 = i1; d1 = d;  i1 = s; }
            } else          { d2 = d;  i2 = s; }
        }
    }
    float w0 = 1.0f / (d0 + 1e-8f);
    float w1 = 1.0f / (d1 + 1e-8f);
    float w2 = 1.0f / (d2 + 1e-8f);
    float ws = w0 + w1 + w2;
    w0 /= ws; w1 /= ws; w2 /= ws;

    size_t base = ((size_t)b * NPTS + n) * 3;
    idx3[base] = i0; idx3[base + 1] = i1; idx3[base + 2] = i2;
    w3[base] = w0;   w3[base + 1] = w1;   w3[base + 2] = w2;
}

// ---------------------------------------------------------------------------
// Kernel 2: build A1 row per point: [p1(3) | interp(768) | zeros(29)] as bf16.
// ---------------------------------------------------------------------------
__global__ __launch_bounds__(256)
void build_a_kernel(const float* __restrict__ points1,
                    const float* __restrict__ points2,
                    const int* __restrict__ idx3, const float* __restrict__ w3,
                    __bf16* __restrict__ A1) {
    const int b = blockIdx.y;
    const int n = blockIdx.x * 256 + threadIdx.x;
    const size_t m = (size_t)b * NPTS + n;

    const int   i0 = idx3[m * 3], i1 = idx3[m * 3 + 1], i2 = idx3[m * 3 + 2];
    const float w0 = w3[m * 3],   w1 = w3[m * 3 + 1],   w2 = w3[m * 3 + 2];

    __bf16* row = A1 + m * K1_PAD;
    const float* p1 = points1 + (size_t)b * 3 * NPTS;
    row[0] = (__bf16)p1[n];
    row[1] = (__bf16)p1[NPTS + n];
    row[2] = (__bf16)p1[2 * NPTS + n];

    const float* p2 = points2 + (size_t)b * C_OUT * SPTS;
    for (int c = 0; c < C_OUT; ++c) {
        const float* pc = p2 + (size_t)c * SPTS;   // 4 KB row: cache friendly
        float v = w0 * pc[i0] + w1 * pc[i1] + w2 * pc[i2];
        row[3 + c] = (__bf16)v;
    }
#pragma unroll
    for (int k = C_IN; k < K1_PAD; ++k) row[k] = (__bf16)0.0f;
}

// ---------------------------------------------------------------------------
// Kernel 3: GEMM1  h[m][j] = relu(alpha1[j] * (A1[m][:] . W1[j][:]) + beta1[j])
// Wave tile: 16 rows x 128 cols (8 accumulators). Per k-chunk: preload all 8
// B-fragments into a distinct-register array (one big load clause), then run
// the 8-WMMA burst with graduated s_wait_loadcnt (one stall per chunk instead
// of one per WMMA). Next-A load is clamped (branch-free).
// grid = (M_TOT/128, HID/128)
// ---------------------------------------------------------------------------
__global__ __launch_bounds__(256)
void gemm1_kernel(const __bf16* __restrict__ A, const __bf16* __restrict__ W,
                  const float* __restrict__ alpha, const float* __restrict__ beta,
                  __bf16* __restrict__ H) {
    const int lane  = threadIdx.x & 31;
    const int wave  = threadIdx.x >> 5;
    const int m0    = blockIdx.x * 128 + wave * 16;
    const int j0    = blockIdx.y * 128;
    const int lrow  = lane & 15;
    const int khalf = (lane >> 4) * 8;
    const int KC    = K1_PAD / 32;                 // 25

    v8f acc[8];
#pragma unroll
    for (int jt = 0; jt < 8; ++jt)
#pragma unroll
        for (int r = 0; r < 8; ++r) acc[jt][r] = 0.0f;

    const __bf16* arow  = A + (size_t)(m0 + lrow) * K1_PAD + khalf;
    const __bf16* wbase = W + (size_t)(j0 + lrow) * K1_PAD + khalf;

    v16bf a = load_frag(arow);
#pragma unroll 1
    for (int kc = 0; kc < KC; ++kc) {
        const int k = kc * 32;
        v16bf bf[8];
#pragma unroll
        for (int jt = 0; jt < 8; ++jt)
            bf[jt] = load_frag(wbase + (size_t)jt * 16 * K1_PAD + k);
        const int kn = (kc + 1 < KC) ? (kc + 1) * 32 : kc * 32;  // clamped
        v16bf a_next = load_frag(arow + kn);
#pragma unroll
        for (int jt = 0; jt < 8; ++jt)
            acc[jt] = __builtin_amdgcn_wmma_f32_16x16x32_bf16(
                false, a, false, bf[jt], (short)0, acc[jt], false, false);
        a = a_next;
    }

#pragma unroll
    for (int jt = 0; jt < 8; ++jt) {
        const int jcol = j0 + jt * 16 + lrow;
        const float sa = alpha[jcol], sb = beta[jcol];
#pragma unroll
        for (int r = 0; r < 8; ++r) {
            const int m = m0 + r + ((lane >> 4) << 3);
            float v = acc[jt][r] * sa + sb;
            v = v > 0.0f ? v : 0.0f;
            H[(size_t)m * HID + jcol] = (__bf16)v;
        }
    }
}

// ---------------------------------------------------------------------------
// Kernel 4: GEMM2 with swapped orientation:
//   D[j][n] = relu(alpha2[j] * (W2[j][:] . h[n][:]) + beta2[j])
// A-operand = W2 rows (M=channels), B-operand = h rows loaded column-major
// (lane = point). D-tile N-dim = points -> contiguous f32 stores to [B,768,N].
// Same preloaded-B-array pipeline as gemm1.
// grid = (M_TOT/128, C_OUT/128)
// ---------------------------------------------------------------------------
__global__ __launch_bounds__(256)
void gemm2_kernel(const __bf16* __restrict__ H, const __bf16* __restrict__ W2b,
                  const float* __restrict__ alpha, const float* __restrict__ beta,
                  float* __restrict__ out) {
    const int lane  = threadIdx.x & 31;
    const int wave  = threadIdx.x >> 5;
    const int n0    = blockIdx.x * 128;
    const int j0    = blockIdx.y * 128 + wave * 16;
    const int lrow  = lane & 15;
    const int khalf = (lane >> 4) * 8;
    const int KC    = HID / 32;                    // 96

    v8f acc[8];
#pragma unroll
    for (int nt = 0; nt < 8; ++nt)
#pragma unroll
        for (int r = 0; r < 8; ++r) acc[nt][r] = 0.0f;

    const __bf16* wrow  = W2b + (size_t)(j0 + lrow) * HID + khalf;
    const __bf16* hbase = H   + (size_t)(n0 + lrow) * HID + khalf;

    v16bf a = load_frag(wrow);
#pragma unroll 1
    for (int kc = 0; kc < KC; ++kc) {
        const int k = kc * 32;
        v16bf bf[8];
#pragma unroll
        for (int nt = 0; nt < 8; ++nt)
            bf[nt] = load_frag(hbase + (size_t)nt * 16 * HID + k);
        const int kn = (kc + 1 < KC) ? (kc + 1) * 32 : kc * 32;  // clamped
        v16bf a_next = load_frag(wrow + kn);
#pragma unroll
        for (int nt = 0; nt < 8; ++nt)
            acc[nt] = __builtin_amdgcn_wmma_f32_16x16x32_bf16(
                false, a, false, bf[nt], (short)0, acc[nt], false, false);
        a = a_next;
    }

    const int bidx  = n0 >> 14;        // batch (128-point tile never crosses b)
    const int nloc0 = n0 & (NPTS - 1);
#pragma unroll
    for (int nt = 0; nt < 8; ++nt) {
        const int n = nloc0 + nt * 16 + lrow;
#pragma unroll
        for (int r = 0; r < 8; ++r) {
            const int j = j0 + r + ((lane >> 4) << 3);
            float v = acc[nt][r] * alpha[j] + beta[j];
            v = v > 0.0f ? v : 0.0f;
            out[((size_t)bidx * C_OUT + j) * NPTS + n] = v;
        }
    }
}

// ---------------------------------------------------------------------------
// Host-side launch
// ---------------------------------------------------------------------------
extern "C" void kernel_launch(void* const* d_in, const int* in_sizes, int n_in,
                              void* d_out, int out_size, void* d_ws, size_t ws_size,
                              hipStream_t stream) {
    const float* xyz1    = (const float*)d_in[0];
    const float* xyz2    = (const float*)d_in[1];
    const float* points1 = (const float*)d_in[2];
    const float* points2 = (const float*)d_in[3];
    const float* W1 = (const float*)d_in[4];
    const float* b1 = (const float*)d_in[5];
    const float* g1 = (const float*)d_in[6];
    const float* be1 = (const float*)d_in[7];
    const float* m1 = (const float*)d_in[8];
    const float* v1 = (const float*)d_in[9];
    const float* W2 = (const float*)d_in[10];
    const float* b2 = (const float*)d_in[11];
    const float* g2 = (const float*)d_in[12];
    const float* be2 = (const float*)d_in[13];
    const float* m2 = (const float*)d_in[14];
    const float* v2 = (const float*)d_in[15];
    float* out = (float*)d_out;
    (void)in_sizes; (void)n_in; (void)out_size; (void)ws_size;

    // Workspace carve-up (all offsets 64B-aligned).
    char* ws = (char*)d_ws;
    size_t off = 0;
    int*    idx3   = (int*)   (ws + off); off += (size_t)M_TOT * 3 * 4;        // 786,432
    float*  w3     = (float*) (ws + off); off += (size_t)M_TOT * 3 * 4;        // 786,432
    __bf16* A1     = (__bf16*)(ws + off); off += (size_t)M_TOT * K1_PAD * 2;   // 100 MB
    __bf16* W1b    = (__bf16*)(ws + off); off += (size_t)HID * K1_PAD * 2;     // 4.9 MB
    __bf16* W2b    = (__bf16*)(ws + off); off += (size_t)C_OUT * HID * 2;      // 4.7 MB
    float*  alpha1 = (float*) (ws + off); off += (size_t)HID * 4;
    float*  beta1  = (float*) (ws + off); off += (size_t)HID * 4;
    float*  alpha2 = (float*) (ws + off); off += (size_t)C_OUT * 4 + 192;
    float*  beta2  = (float*) (ws + off); off += (size_t)C_OUT * 4 + 192;
    __bf16* Hbuf   = (__bf16*)(ws + off); off += (size_t)M_TOT * HID * 2;      // 402 MB

    // 0) BN folding + bf16 weight conversion
    {
        const size_t total = (size_t)HID * K1_PAD + (size_t)C_OUT * HID + HID + C_OUT;
        dim3 grid((unsigned)((total + 255) / 256));
        prep_kernel<<<grid, 256, 0, stream>>>(W1, W2, b1, g1, be1, m1, v1,
                                              b2, g2, be2, m2, v2,
                                              W1b, W2b, alpha1, beta1, alpha2, beta2);
    }
    // 1) 3-NN + weights
    knn_kernel<<<dim3(NPTS / 256, BATCH), 256, 0, stream>>>(xyz1, xyz2, idx3, w3);
    // 2) interpolation + concat -> bf16 A matrix
    build_a_kernel<<<dim3(NPTS / 256, BATCH), 256, 0, stream>>>(points1, points2,
                                                                idx3, w3, A1);
    // 3) GEMM1 (771->3072) + BN + ReLU, bf16 hidden
    gemm1_kernel<<<dim3(M_TOT / 128, HID / 128), 256, 0, stream>>>(A1, W1b,
                                                                   alpha1, beta1, Hbuf);
    // 4) GEMM2 (3072->768) + BN + ReLU, f32 output [B, 768, N]
    gemm2_kernel<<<dim3(M_TOT / 128, C_OUT / 128), 256, 0, stream>>>(Hbuf, W2b,
                                                                     alpha2, beta2, out);
}
